// VectorQuantizer_18794776887415
// MI455X (gfx1250) — compile-verified, matrix operand order checked by reading, not run on hardware
//
#include <hip/hip_runtime.h>
#include <hip/hip_bf16.h>

#define DIMV    512
#define KCB     1024
#define TLEN    2048
#define BSZ     16
#define NROWS   (BSZ * TLEN)          /* 32768 */
#define ZQELEMS (BSZ * DIMV * TLEN)   /* 16777216 */

typedef __attribute__((ext_vector_type(16))) __bf16 v16bf;
typedef __attribute__((ext_vector_type(8)))  float  v8f;

union FragAB { v16bf v; unsigned u[8]; uint4 q[2]; };
union FragC  { v8f   v; float    f[8]; };

__device__ __forceinline__ unsigned short f2bf(float f) {
    unsigned u = __float_as_uint(f);
    u += 0x7FFFu + ((u >> 16) & 1u);        // round-to-nearest-even
    return (unsigned short)(u >> 16);
}

// ---- workspace layout (bytes) ----
#define WS_WSQ_OFF 256       // float w_sq[1024]
#define WS_IDX_OFF 8192      // int   idx[32768]
#define WS_CB_OFF  262144    // ushort cb_bf16[1024*512]  (1 MB)

// Kernel 1: codebook -> bf16, w_sq (exact fp32), zero loss accumulator.
__global__ void vq_prep(const float* __restrict__ cb,
                        float* __restrict__ wsq,
                        unsigned short* __restrict__ cbbf,
                        float* __restrict__ lossAcc) {
    __shared__ float red[256];
    int k   = blockIdx.x;
    int tid = threadIdx.x;
    if (k == 0 && tid == 0) *lossAcc = 0.0f;
    const float* row = cb + (size_t)k * DIMV;
    int d0 = tid * 2;
    float a = row[d0], b = row[d0 + 1];
    cbbf[(size_t)k * DIMV + d0]     = f2bf(a);
    cbbf[(size_t)k * DIMV + d0 + 1] = f2bf(b);
    red[tid] = a * a + b * b;
    __syncthreads();
    for (int off = 128; off > 0; off >>= 1) {
        if (tid < off) red[tid] += red[tid + off];
        __syncthreads();
    }
    if (tid == 0) wsq[k] = red[0];
}

// Kernel 2: argmin via bf16 WMMA. 256 threads = 8 waves, 128 rows/block.
// A tile held in registers; codebook tiles double-buffered in LDS via
// global_load_async_to_lds_b128 (ASYNCcnt).
__global__ void __launch_bounds__(256)
vq_argmin(const float* __restrict__ z,
          const unsigned short* __restrict__ cbbf,
          const float* __restrict__ wsq,
          int* __restrict__ idx_out,
          float* __restrict__ idxf_out) {
    __shared__ unsigned short cbsh[2][16][DIMV];  // 2 x 16 KB double buffer
    __shared__ float wsq_sh[KCB];                 // 4 KB

    int tid  = threadIdx.x;
    int lane = tid & 31;
    int wv   = tid >> 5;
    int n0   = blockIdx.x * 128;
    int b    = n0 >> 11;             // T = 2048
    int t0   = n0 & (TLEN - 1);
    int m    = lane & 15;
    int half = lane >> 4;

    for (int i = tid; i < KCB; i += 256) wsq_sh[i] = wsq[i];

    // ---- preload this wave's A tile (16 rows x 512 d) into registers ----
    // row of lane = t0 + wv*16 + m; for fixed d, 16 lanes hit one 64B segment.
    const float* zrow = z + (size_t)b * (DIMV * TLEN) + (t0 + wv * 16 + m);
    FragAB fa[16];
#pragma unroll
    for (int dc = 0; dc < 16; ++dc) {
#pragma unroll
        for (int p = 0; p < 8; ++p) {
            int d = dc * 32 + ((p < 4) ? (2 * p) : (2 * p + 8)) + half * 8;
            float f0 = zrow[(size_t)d * TLEN];
            float f1 = zrow[(size_t)(d + 1) * TLEN];
            fa[dc].u[p] = ((unsigned)f2bf(f1) << 16) | (unsigned)f2bf(f0);
        }
    }

    // raw LDS byte offset of cbsh (generic shared ptr low 32 bits = LDS offset)
    unsigned cb_lds = (unsigned)(size_t)(void*)&cbsh[0][0][0];

    // async copy of one 16KB codebook tile: 256 thr x 4 x B128
    auto issue_tile = [&](int ct, int buf) {
        const char* gsrc = (const char*)(cbbf + (size_t)ct * 16 * DIMV);
        unsigned base = cb_lds + (unsigned)buf * 16384u;
#pragma unroll
        for (int j = 0; j < 4; ++j) {
            unsigned off = (unsigned)(tid + 256 * j) * 16u;
            asm volatile("global_load_async_to_lds_b128 %0, %1, %2"
                         :: "v"(base + off), "v"(off), "s"(gsrc)
                         : "memory");
        }
    };

    float minv[8];
    int   mini[8];
#pragma unroll
    for (int r = 0; r < 8; ++r) { minv[r] = 3.4e38f; mini[r] = 0; }

    issue_tile(0, 0);

    for (int ct = 0; ct < 64; ++ct) {
        int buf = ct & 1;
        asm volatile("s_wait_asynccnt 0x0" ::: "memory");
        __syncthreads();                       // tile ct ready; prev readers done
        if (ct + 1 < 64) issue_tile(ct + 1, buf ^ 1);

        float wsq_l = wsq_sh[ct * 16 + m];
        FragC acc;
#pragma unroll
        for (int r = 0; r < 8; ++r) acc.f[r] = 0.0f;

        const uint4* brow = (const uint4*)&cbsh[buf][m][0];
#pragma unroll
        for (int dc = 0; dc < 16; ++dc) {
            FragAB fb;
            fb.q[0] = brow[dc * 4 + half];
            fb.q[1] = brow[dc * 4 + 2 + half];
            acc.v = __builtin_amdgcn_wmma_f32_16x16x32_bf16(
                false, fa[dc].v, false, fb.v, (short)0, acc.v, false, false);
        }

        int cidx = ct * 16 + m;
#pragma unroll
        for (int r = 0; r < 8; ++r) {
            float dist = fmaf(-2.0f, acc.f[r], wsq_l);   // |w|^2 - 2 z.w
            if (dist < minv[r]) { minv[r] = dist; mini[r] = cidx; }
        }
    }

    // Min-reduce (value, index) across the 16 lanes of each half.
#pragma unroll
    for (int off = 1; off < 16; off <<= 1) {
#pragma unroll
        for (int r = 0; r < 8; ++r) {
            float ov = __shfl_xor(minv[r], off, 32);
            int   oi = __shfl_xor(mini[r], off, 32);
            if (ov < minv[r] || (ov == minv[r] && oi < mini[r])) {
                minv[r] = ov; mini[r] = oi;
            }
        }
    }
    if (m == 0) {
        int rowbase = n0 + wv * 16 + half * 8;   // VGPR r -> row half*8+r
#pragma unroll
        for (int r = 0; r < 8; ++r) {
            idx_out[rowbase + r]  = mini[r];
            idxf_out[rowbase + r] = (float)mini[r];
        }
    }
}

// Kernel 3: gather z_q = codebook[idx] into [B,D,T] + squared-error partial sums.
__global__ void vq_gather(const float* __restrict__ z,
                          const float* __restrict__ cb,
                          const int* __restrict__ idx,
                          float* __restrict__ zq_out,
                          float* __restrict__ lossAcc) {
    __shared__ float red[256];
    int e = blockIdx.x * 256 + threadIdx.x;
    int t = e & (TLEN - 1);
    int d = (e >> 11) & (DIMV - 1);
    int b = e >> 20;
    int n = (b << 11) + t;
    int ci = idx[n];
    float v = cb[(size_t)ci * DIMV + d];
    zq_out[e] = v;
    float diff = v - z[e];
    red[threadIdx.x] = diff * diff;
    __syncthreads();
    for (int off = 128; off > 0; off >>= 1) {
        if (threadIdx.x < off) red[threadIdx.x] += red[threadIdx.x + off];
        __syncthreads();
    }
    if (threadIdx.x == 0) atomicAdd(lossAcc, red[0]);
}

// Kernel 4: loss = (1 + beta) * sum / (N*D), beta = 0.25
__global__ void vq_final(const float* __restrict__ lossAcc,
                         float* __restrict__ out_loss) {
    if (threadIdx.x == 0)
        *out_loss = (*lossAcc) * 1.25f / (float)ZQELEMS;
}

extern "C" void kernel_launch(void* const* d_in, const int* in_sizes, int n_in,
                              void* d_out, int out_size, void* d_ws, size_t ws_size,
                              hipStream_t stream) {
    const float* z  = (const float*)d_in[0];
    const float* cb = (const float*)d_in[1];
    float* out = (float*)d_out;
    char*  ws  = (char*)d_ws;

    float*          lossAcc = (float*)ws;
    float*          wsq     = (float*)(ws + WS_WSQ_OFF);
    int*            idxw    = (int*)(ws + WS_IDX_OFF);
    unsigned short* cbbf    = (unsigned short*)(ws + WS_CB_OFF);

    float* zq_out   = out;                       // [B, D, T] = 16,777,216
    float* idx_f    = out + ZQELEMS;             // [B, T]    = 32,768
    float* loss_out = out + ZQELEMS + NROWS;     // scalar

    vq_prep  <<<KCB,           256, 0, stream>>>(cb, wsq, cbbf, lossAcc);
    vq_argmin<<<NROWS / 128,   256, 0, stream>>>(z, cbbf, wsq, idxw, idx_f);
    vq_gather<<<ZQELEMS / 256, 256, 0, stream>>>(z, cb, idxw, zq_out, lossAcc);
    vq_final <<<1,               1, 0, stream>>>(lossAcc, loss_out);
}
